// RecurrentBattleNet_21766894256155
// MI455X (gfx1250) — matrix operand order repaired and checked by reading, not compiled, not run on hardware
//
#include <hip/hip_runtime.h>
#include <hip/hip_bf16.h>

// ---------------------------------------------------------------------------
// RecurrentBattleNet for MI455X (gfx1250): f16 WMMA GEMMs + fused LSTM scan.
//   INPUT=512, HIDDEN=128 (GATES=512), LAYERS=2, B=512, T=256
// Phases:
//   1) convert: f32 -> f16 copies of x and all LSTM weights; fold biases.
//   2) xp0 = x @ Wih0^T + (bih0+bhh0)   (WMMA f16, M=131072,N=512,K=512)
//   3) fused 2-layer LSTM scan: 32 blocks x 16 batch rows, 8 waves split the
//      512 gate columns, WMMA K=128 chains per step, h exchanged via LDS.
//   4) FC head + softmax (tiny, scalar).
// ---------------------------------------------------------------------------

typedef __attribute__((ext_vector_type(16))) _Float16     v16h;
typedef __attribute__((ext_vector_type(8)))  float        v8f;
typedef __attribute__((ext_vector_type(4)))  unsigned int v4u;

#define INPUT_D 512
#define HIDDEN_D 128
#define GATES_D 512
#define B_SZ 512
#define T_SZ 256
#define M_ROWS (B_SZ * T_SZ)   // 131072

union Frag {
    v16h h;
    v4u  q[2];
};

// A fragment, 16x32 f16, from row-major f16 (ld elements per row).
// base points at (row0, k0). Lane layout per CDNA5 ISA 7.12.2:
//   lanes 0-15:  M=lane,    K = {k0..k0+7} U {k0+16..k0+23}
//   lanes 16-31: M=lane-16, K = {k0+8..k0+15} U {k0+24..k0+31}
__device__ __forceinline__ v16h load_a16x32(const _Float16* base, int ld, int lane) {
    const int half = lane >> 4;
    const int m    = lane & 15;
    const _Float16* p = base + m * ld + half * 8;
    Frag f;
    f.q[0] = *(const v4u*)(p);        // 8 halves
    f.q[1] = *(const v4u*)(p + 16);   // 8 halves, +16 columns
    return f.h;
}

// B fragment, 32x16 f16, B[k][n] = W[n][k] with W row-major [N][K].
// wbase points at (n0, k0). Lane = column n0+(lane&15);
//   lanes 0-15 carry K = {k0..k0+7} U {k0+16..k0+23}, lanes 16-31 carry K = {k0+8..k0+15} U {k0+24..k0+31}.
__device__ __forceinline__ v16h load_b32x16(const _Float16* wbase, int ld, int lane) {
    const int half = lane >> 4;
    const int n    = lane & 15;
    const _Float16* p = wbase + n * ld + half * 8;
    Frag f;
    f.q[0] = *(const v4u*)(p);
    f.q[1] = *(const v4u*)(p + 16);
    return f.h;
}

__device__ __forceinline__ v8f wmma_f16(v16h a, v16h b, v8f c) {
    // D = A x B + C, f32 accumulate. (neg_a, A, neg_b, B, c_mod, C, reuse_a, reuse_b)
    return __builtin_amdgcn_wmma_f32_16x16x32_f16(false, a, false, b, (short)0, c, false, false);
}

__device__ __forceinline__ float sigf(float x) { return 1.0f / (1.0f + __expf(-x)); }
__device__ __forceinline__ float tanhfast(float x) { return 1.0f - 2.0f / (__expf(2.0f * x) + 1.0f); }

// ---------------------------------------------------------------------------
// Phase 1: precision conversion + bias folding
// ---------------------------------------------------------------------------
__global__ void k_convert(const float* __restrict__ x,
                          const float* __restrict__ Wih0, const float* __restrict__ Whh0,
                          const float* __restrict__ bih0, const float* __restrict__ bhh0,
                          const float* __restrict__ Wih1, const float* __restrict__ Whh1,
                          const float* __restrict__ bih1, const float* __restrict__ bhh1,
                          _Float16* __restrict__ x16,
                          _Float16* __restrict__ wih0_16, _Float16* __restrict__ whh0_16,
                          _Float16* __restrict__ wih1_16, _Float16* __restrict__ whh1_16,
                          float* __restrict__ bias0, float* __restrict__ bias1) {
    const long long NX = (long long)M_ROWS * INPUT_D;
    const long long stride = (long long)gridDim.x * blockDim.x;
    for (long long i = (long long)blockIdx.x * blockDim.x + threadIdx.x; i < NX; i += stride)
        x16[i] = (_Float16)x[i];

    const int t = blockIdx.x * blockDim.x + threadIdx.x;
    const int sstep = gridDim.x * blockDim.x;
    for (int j = t; j < GATES_D * INPUT_D; j += sstep) wih0_16[j] = (_Float16)Wih0[j];
    for (int j = t; j < GATES_D * HIDDEN_D; j += sstep) {
        whh0_16[j] = (_Float16)Whh0[j];
        wih1_16[j] = (_Float16)Wih1[j];
        whh1_16[j] = (_Float16)Whh1[j];
    }
    for (int j = t; j < GATES_D; j += sstep) {
        bias0[j] = bih0[j] + bhh0[j];
        bias1[j] = bih1[j] + bhh1[j];
    }
}

// ---------------------------------------------------------------------------
// Phase 2: xp0[m][g] = sum_k x16[m][k] * Wih0[g][k] + bias0[g]   (f16 out)
// One wave computes a 16x64 output slab (A fragment reused over 4 N-tiles).
// ---------------------------------------------------------------------------
__global__ void k_gemm_xp0(const _Float16* __restrict__ x16,
                           const _Float16* __restrict__ w16,   // [512][512] row-major
                           const float* __restrict__ bias0,
                           _Float16* __restrict__ xp0) {
    const int lane = threadIdx.x & 31;
    const int wid  = threadIdx.x >> 5;
    const int gw   = blockIdx.x * 8 + wid;       // global wave id
    const int ngrp = GATES_D / 64;               // 8 groups of 64 columns
    const int mtile = gw / ngrp;
    const int ngr   = gw % ngrp;
    const int row0  = mtile * 16;
    const int n0    = ngr * 64;

    const int half = lane >> 4;
    const int nl   = lane & 15;

    v8f acc[4];
#pragma unroll
    for (int i = 0; i < 4; ++i) {
        const float bv = bias0[n0 + i * 16 + nl];
#pragma unroll
        for (int r = 0; r < 8; ++r) acc[i][r] = bv;
    }

    const _Float16* arow = x16 + (long long)row0 * INPUT_D;
    for (int kb = 0; kb < INPUT_D / 32; ++kb) {
        const int k0 = kb * 32;
        v16h a = load_a16x32(arow + k0, INPUT_D, lane);
        if (kb + 1 < INPUT_D / 32)
            __builtin_prefetch(arow + k0 + 32, 0, 1);   // global_prefetch_b8
#pragma unroll
        for (int i = 0; i < 4; ++i) {
            v16h b = load_b32x16(w16 + (long long)(n0 + i * 16) * INPUT_D + k0, INPUT_D, lane);
            acc[i] = wmma_f16(a, b, acc[i]);
        }
    }

#pragma unroll
    for (int i = 0; i < 4; ++i)
#pragma unroll
        for (int r = 0; r < 8; ++r) {
            const int m = r + 8 * half;
            xp0[(long long)(row0 + m) * GATES_D + n0 + i * 16 + nl] = (_Float16)acc[i][r];
        }
}

// ---------------------------------------------------------------------------
// Phase 3: fused 2-layer LSTM scan.
// Block bb owns batch rows [16*bb, 16*bb+16). 8 waves; wave j owns hidden
// columns [16*j, 16*j+16) i.e. gate columns {q*128 + 16*j + n} for q=i,f,g,o.
// Per step: gates0 = xp0[.,t,.] + h0 @ Whh0^T ; LSTM update -> h0[t]
//           gates1 = bias1 + h0[t] @ Wih1^T + h1 @ Whh1^T ; update -> h1[t]
// h exchanged through LDS (ds_load_b128 A-fragments), 3 barriers / step.
// ---------------------------------------------------------------------------
__global__ void k_recurrent(const _Float16* __restrict__ xp0,   // [B*T][512], row = b*T + t
                            const _Float16* __restrict__ whh0,  // [512][128]
                            const _Float16* __restrict__ wih1,  // [512][128]
                            const _Float16* __restrict__ whh1,  // [512][128]
                            const float* __restrict__ bias1,    // [512]
                            float* __restrict__ out,            // d_out
                            float* __restrict__ h1_last) {      // [B][128] f32 scratch
    __shared__ _Float16 h0s[16 * HIDDEN_D];
    __shared__ _Float16 h1s[16 * HIDDEN_D];

    const int lane = threadIdx.x & 31;
    const int wv   = threadIdx.x >> 5;    // 0..7 -> hidden chunk
    const int b0   = blockIdx.x * 16;
    const int half = lane >> 4;
    const int nl   = lane & 15;
    const int jc   = wv * 16;             // hidden column base for this wave

    for (int i = threadIdx.x; i < 16 * HIDDEN_D; i += blockDim.x) {
        h0s[i] = (_Float16)0.f;
        h1s[i] = (_Float16)0.f;
    }
    __syncthreads();

    v8f c0 = {0.f, 0.f, 0.f, 0.f, 0.f, 0.f, 0.f, 0.f};
    v8f c1 = {0.f, 0.f, 0.f, 0.f, 0.f, 0.f, 0.f, 0.f};

    float* h_n = out + B_SZ * 10;                 // [2][B][H]
    float* c_n = h_n + 2 * B_SZ * HIDDEN_D;       // [2][B][H]

    for (int t = 0; t < T_SZ; ++t) {
        // --- read h[t-1] fragments (all waves read full 16x128 tiles) ---
        v16h a0[4], a1[4];
#pragma unroll
        for (int kb = 0; kb < 4; ++kb) {
            a0[kb] = load_a16x32(h0s + kb * 32, HIDDEN_D, lane);
            a1[kb] = load_a16x32(h1s + kb * 32, HIDDEN_D, lane);
        }
        __syncthreads();   // everyone is done reading h[t-1]

        // --- layer 0 gates for this wave's 16 hidden columns ---
        v8f g[4];
#pragma unroll
        for (int q = 0; q < 4; ++q) {
            const int gc = q * HIDDEN_D + jc + nl;
            v8f acc;
#pragma unroll
            for (int r = 0; r < 8; ++r) {
                const int b = b0 + r + 8 * half;
                acc[r] = (float)xp0[((long long)b * T_SZ + t) * GATES_D + gc];
            }
#pragma unroll
            for (int kb = 0; kb < 4; ++kb) {
                v16h bf = load_b32x16(whh0 + (q * HIDDEN_D + jc) * HIDDEN_D + kb * 32,
                                      HIDDEN_D, lane);
                acc = wmma_f16(a0[kb], bf, acc);
            }
            g[q] = acc;
        }
        v8f h0new;
#pragma unroll
        for (int r = 0; r < 8; ++r) {
            const float iv = sigf(g[0][r]);
            const float fv = sigf(g[1][r]);
            const float gv = tanhfast(g[2][r]);
            const float ov = sigf(g[3][r]);
            const float cc = fv * c0[r] + iv * gv;
            c0[r] = cc;
            h0new[r] = ov * tanhfast(cc);
        }
#pragma unroll
        for (int r = 0; r < 8; ++r)
            h0s[(r + 8 * half) * HIDDEN_D + jc + nl] = (_Float16)h0new[r];
        __syncthreads();   // h0s now holds h0[t]

        // --- layer 1: input projection of h0[t] plus recurrent term ---
        v16h ax[4];
#pragma unroll
        for (int kb = 0; kb < 4; ++kb)
            ax[kb] = load_a16x32(h0s + kb * 32, HIDDEN_D, lane);

#pragma unroll
        for (int q = 0; q < 4; ++q) {
            const int gc = q * HIDDEN_D + jc + nl;
            const float bv = bias1[gc];
            v8f acc;
#pragma unroll
            for (int r = 0; r < 8; ++r) acc[r] = bv;
#pragma unroll
            for (int kb = 0; kb < 4; ++kb) {
                v16h bfi = load_b32x16(wih1 + (q * HIDDEN_D + jc) * HIDDEN_D + kb * 32,
                                       HIDDEN_D, lane);
                acc = wmma_f16(ax[kb], bfi, acc);
                v16h bfh = load_b32x16(whh1 + (q * HIDDEN_D + jc) * HIDDEN_D + kb * 32,
                                       HIDDEN_D, lane);
                acc = wmma_f16(a1[kb], bfh, acc);
            }
            g[q] = acc;
        }
        v8f h1new;
#pragma unroll
        for (int r = 0; r < 8; ++r) {
            const float iv = sigf(g[0][r]);
            const float fv = sigf(g[1][r]);
            const float gv = tanhfast(g[2][r]);
            const float ov = sigf(g[3][r]);
            const float cc = fv * c1[r] + iv * gv;
            c1[r] = cc;
            h1new[r] = ov * tanhfast(cc);
        }
#pragma unroll
        for (int r = 0; r < 8; ++r)
            h1s[(r + 8 * half) * HIDDEN_D + jc + nl] = (_Float16)h1new[r];

        if (t == T_SZ - 1) {
#pragma unroll
            for (int r = 0; r < 8; ++r) {
                const int b   = b0 + r + 8 * half;
                const int col = jc + nl;
                h_n[0 * B_SZ * HIDDEN_D + b * HIDDEN_D + col] = h0new[r];
                h_n[1 * B_SZ * HIDDEN_D + b * HIDDEN_D + col] = h1new[r];
                c_n[0 * B_SZ * HIDDEN_D + b * HIDDEN_D + col] = c0[r];
                c_n[1 * B_SZ * HIDDEN_D + b * HIDDEN_D + col] = c1[r];
                h1_last[b * HIDDEN_D + col] = h1new[r];
            }
        }
        __syncthreads();   // h1s writes visible to next iteration's reads
    }
}

// ---------------------------------------------------------------------------
// Phase 4: FC head + softmax (one thread per batch row; trivially small)
// ---------------------------------------------------------------------------
__global__ void k_fc(const float* __restrict__ h1_last,
                     const float* __restrict__ fc1_w, const float* __restrict__ fc1_b,
                     const float* __restrict__ fc2_w, const float* __restrict__ fc2_b,
                     float* __restrict__ probs) {
    const int b = blockIdx.x * blockDim.x + threadIdx.x;
    if (b >= B_SZ) return;
    const float* hrow = h1_last + b * HIDDEN_D;
    float hid[64];
    for (int o = 0; o < 64; ++o) {
        float s = fc1_b[o];
        const float* w = fc1_w + o * HIDDEN_D;
        for (int k = 0; k < HIDDEN_D; ++k) s += hrow[k] * w[k];
        hid[o] = fmaxf(s, 0.f);
    }
    float lg[10];
    float mx = -3.0e38f;
    for (int o = 0; o < 10; ++o) {
        float s = fc2_b[o];
        const float* w = fc2_w + o * 64;
        for (int k = 0; k < 64; ++k) s += hid[k] * w[k];
        lg[o] = s;
        mx = fmaxf(mx, s);
    }
    float sum = 0.f;
    for (int o = 0; o < 10; ++o) { lg[o] = __expf(lg[o] - mx); sum += lg[o]; }
    const float inv = 1.f / sum;
    for (int o = 0; o < 10; ++o) probs[b * 10 + o] = lg[o] * inv;
}

// ---------------------------------------------------------------------------
extern "C" void kernel_launch(void* const* d_in, const int* in_sizes, int n_in,
                              void* d_out, int out_size, void* d_ws, size_t ws_size,
                              hipStream_t stream) {
    const float* x    = (const float*)d_in[0];
    const float* Wih0 = (const float*)d_in[1];
    const float* Whh0 = (const float*)d_in[2];
    const float* bih0 = (const float*)d_in[3];
    const float* bhh0 = (const float*)d_in[4];
    const float* Wih1 = (const float*)d_in[5];
    const float* Whh1 = (const float*)d_in[6];
    const float* bih1 = (const float*)d_in[7];
    const float* bhh1 = (const float*)d_in[8];
    const float* fc1w = (const float*)d_in[9];
    const float* fc1b = (const float*)d_in[10];
    const float* fc2w = (const float*)d_in[11];
    const float* fc2b = (const float*)d_in[12];
    float* out = (float*)d_out;

    // workspace layout (all offsets 256B-aligned by construction)
    char* ws = (char*)d_ws;
    size_t cur = 0;
    _Float16* x16   = (_Float16*)(ws + cur); cur += (size_t)M_ROWS * INPUT_D * 2;   // 134 MB
    _Float16* xp0   = (_Float16*)(ws + cur); cur += (size_t)M_ROWS * GATES_D * 2;   // 134 MB
    _Float16* wih0h = (_Float16*)(ws + cur); cur += (size_t)GATES_D * INPUT_D * 2;  // 512 KB
    _Float16* whh0h = (_Float16*)(ws + cur); cur += (size_t)GATES_D * HIDDEN_D * 2; // 128 KB
    _Float16* wih1h = (_Float16*)(ws + cur); cur += (size_t)GATES_D * HIDDEN_D * 2;
    _Float16* whh1h = (_Float16*)(ws + cur); cur += (size_t)GATES_D * HIDDEN_D * 2;
    float*    bias0 = (float*)(ws + cur);    cur += (size_t)GATES_D * 4;
    float*    bias1 = (float*)(ws + cur);    cur += (size_t)GATES_D * 4;
    float*    h1l   = (float*)(ws + cur);    cur += (size_t)B_SZ * HIDDEN_D * 4;

    k_convert<<<2048, 256, 0, stream>>>(x, Wih0, Whh0, bih0, bhh0, Wih1, Whh1, bih1, bhh1,
                                        x16, wih0h, whh0h, wih1h, whh1h, bias0, bias1);

    // 8192 M-tiles x 8 N-groups = 65536 waves = 8192 blocks of 8 waves
    k_gemm_xp0<<<8192, 256, 0, stream>>>(x16, wih0h, bias0, xp0);

    // 32 blocks x 16 batch rows, 8 waves each
    k_recurrent<<<B_SZ / 16, 256, 0, stream>>>(xp0, whh0h, wih1h, whh1h, bias1, out, h1l);

    k_fc<<<2, 256, 0, stream>>>(h1l, fc1w, fc1b, fc2w, fc2b, out);
}
